// LoRALinear_17635135717659
// MI455X (gfx1250) — matrix operand order repaired and checked by reading, not compiled, not run on hardware
//
#include <hip/hip_runtime.h>

// ---- problem constants (from reference) ----
constexpr int D_K = 4096;     // D_IN
constexpr int D_N = 4096;     // D_OUT
constexpr int RANK = 16;
constexpr float SCALE = 2.0f; // ALPHA / RANK = 32/16

// ---- tiling ----
constexpr int BM = 128, BN = 128, BK = 64;
constexpr int PITCH = BK + 8;      // bf16 elems per LDS row (144 B, breaks bank conflicts)
constexpr int NTHREADS = 256;      // 8 waves (wave32)

typedef __bf16        v16bf __attribute__((ext_vector_type(16)));
typedef float         v8f   __attribute__((ext_vector_type(8)));
typedef float         f32x4 __attribute__((ext_vector_type(4)));
typedef unsigned int  u32x4 __attribute__((ext_vector_type(4)));
typedef unsigned int  u32x2 __attribute__((ext_vector_type(2)));
typedef int           i32x4 __attribute__((ext_vector_type(4)));

// ---------------- helpers ----------------

// pack 4 fp32 -> 4 bf16 (two dwords) ; compiler emits v_cvt_pk_bf16_f32
__device__ __forceinline__ u32x2 cvt_bf16x4(f32x4 f) {
    unsigned short a = __builtin_bit_cast(unsigned short, (__bf16)f.x);
    unsigned short b = __builtin_bit_cast(unsigned short, (__bf16)f.y);
    unsigned short c = __builtin_bit_cast(unsigned short, (__bf16)f.z);
    unsigned short d = __builtin_bit_cast(unsigned short, (__bf16)f.w);
    u32x2 r;
    r.x = (unsigned)a | ((unsigned)b << 16);
    r.y = (unsigned)c | ((unsigned)d << 16);
    return r;
}

// 16-byte async DMA global -> LDS (ASYNCcnt-tracked, no VGPR staging)
__device__ __forceinline__ void async_copy16(const void* g, void* lds) {
#if __has_builtin(__builtin_amdgcn_global_load_async_to_lds_b128)
    // signature (probe-confirmed): (int4 AS1* src, int4 AS3* dst, i32 offset, i32 cpol)
    __builtin_amdgcn_global_load_async_to_lds_b128(
        (__attribute__((address_space(1))) i32x4*)(uintptr_t)g,
        (__attribute__((address_space(3))) i32x4*)lds, 0, 0);
#else
    unsigned lo = (unsigned)(uintptr_t)(__attribute__((address_space(3))) char*)lds;
    asm volatile("global_load_async_to_lds_b128 %0, %1, off"
                 :: "v"(lo), "v"((unsigned long long)(uintptr_t)g) : "memory");
#endif
}

__device__ __forceinline__ void wait_async0() {
#if __has_builtin(__builtin_amdgcn_s_wait_asynccnt)
    __builtin_amdgcn_s_wait_asynccnt(0);
#else
    asm volatile("s_wait_asynccnt 0x0" ::: "memory");
#endif
}

// cooperative 128x64 fp32 tile load: each of 256 threads grabs 8 float4s
__device__ __forceinline__ void load_stage(const float* __restrict__ g, int row0,
                                           int k0, f32x4 st[8], int tid) {
    int r  = tid >> 4;
    int c4 = tid & 15;
#pragma unroll
    for (int i = 0; i < 8; ++i)
        st[i] = *(const f32x4*)(g + (size_t)(row0 + r + i * 16) * D_K + k0 + c4 * 4);
}

// convert + store staged tile into LDS (bf16, padded pitch)
__device__ __forceinline__ void store_stage(__bf16* s, const f32x4 st[8], int tid) {
    int r  = tid >> 4;
    int c4 = tid & 15;
#pragma unroll
    for (int i = 0; i < 8; ++i)
        *(u32x2*)&s[(r + i * 16) * PITCH + c4 * 4] = cvt_bf16x4(st[i]);
}

// async-DMA a 128xBK bf16 tile (already-converted global data) into padded LDS
__device__ __forceinline__ void async_tile(const __bf16* __restrict__ g, int row0,
                                           int k0, __bf16* lds, int tid) {
#pragma unroll
    for (int i = 0; i < 4; ++i) {
        int c   = tid + i * NTHREADS;   // 0..1023 : 128 rows x 8 chunks of 8 bf16
        int row = c >> 3, ch = c & 7;
        const __bf16* gp = g + (size_t)(row0 + row) * D_K + k0 + ch * 8;
        __bf16*       lp = lds + row * PITCH + ch * 8;
        async_copy16(gp, lp);
    }
}

// A fragment (16x32 bf16): lane<16 -> K {ks..+7, ks+16..+23}; lane>=16 -> +8
__device__ __forceinline__ v16bf load_fragA(const __bf16* s, int row, int ks, int lane) {
    int r  = row + (lane & 15);
    int kh = ks + ((lane >> 4) << 3);
    union { v16bf v; u32x4 q[2]; } f;
    const __bf16* p = s + r * PITCH + kh;
    f.q[0] = *(const u32x4*)(p);
    f.q[1] = *(const u32x4*)(p + 16);
    return f.v;
}

// B fragment (32x16 bf16, col n per lane): lane<16 -> K {ks..+7, ks+16..+23}; lane>=16 -> +8
__device__ __forceinline__ v16bf load_fragB(const __bf16* s, int row, int ks, int lane) {
    int r  = row + (lane & 15);
    int kh = ks + ((lane >> 4) << 3);
    union { v16bf v; u32x4 q[2]; } f;
    const __bf16* p = s + r * PITCH + kh;
    f.q[0] = *(const u32x4*)(p);
    f.q[1] = *(const u32x4*)(p + 16);
    return f.v;
}

// shared epilogue: out = acc + bias[n] + SCALE * XA[m,:] . B[n,:]
__device__ __forceinline__ void epilogue(v8f acc[2][4], int m0, int n0, int wm, int wn,
                                         int lane, const float* __restrict__ bias,
                                         const float* __restrict__ loraB,
                                         const float* __restrict__ xa,
                                         float* __restrict__ out) {
    int lm = lane & 15;
    int lh = (lane >> 4) << 3;           // C-frag: upper lane half -> M+8
#pragma unroll
    for (int tj = 0; tj < 4; ++tj) {
        int n = n0 + wn + tj * 16 + lm;
        float bn = bias[n];
        const f32x4* bp = (const f32x4*)(loraB + (size_t)n * RANK);
        f32x4 b0 = bp[0], b1 = bp[1], b2 = bp[2], b3 = bp[3];
#pragma unroll
        for (int ti = 0; ti < 2; ++ti) {
#pragma unroll
            for (int j = 0; j < 8; ++j) {
                int m = m0 + wm + ti * 16 + lh + j;
                const f32x4* xp = (const f32x4*)(xa + (size_t)m * RANK);
                f32x4 x0 = xp[0], x1 = xp[1], x2 = xp[2], x3 = xp[3];
                float s = x0.x * b0.x + x0.y * b0.y + x0.z * b0.z + x0.w * b0.w
                        + x1.x * b1.x + x1.y * b1.y + x1.z * b1.z + x1.w * b1.w
                        + x2.x * b2.x + x2.y * b2.y + x2.z * b2.z + x2.w * b2.w
                        + x3.x * b3.x + x3.y * b3.y + x3.z * b3.z + x3.w * b3.w;
                out[(size_t)m * D_N + n] = acc[ti][tj][j] + bn + SCALE * s;
            }
        }
    }
}

// ---------------- stage 0: fp32 -> bf16 bulk convert ----------------
__global__ __launch_bounds__(256) void cvt_bf16_kernel(const float* __restrict__ in,
                                                       u32x2* __restrict__ out, int n4) {
    int i = blockIdx.x * 256 + threadIdx.x;
    if (i < n4) out[i] = cvt_bf16x4(((const f32x4*)in)[i]);
}

// ---------------- stage 1: XA[m, r] = sum_k X[m,k] * A[r,k] ----------------
__global__ __launch_bounds__(256) void lora_xa_kernel(const float* __restrict__ x,
                                                      const float* __restrict__ loraA,
                                                      float* __restrict__ xa) {
    __shared__ float sx[16][129];
    __shared__ float sa[16][129];
    int m0  = blockIdx.x * 16;
    int tid = threadIdx.x;
    int mi = tid >> 4, r = tid & 15;
    float acc = 0.f;
    for (int k0 = 0; k0 < D_K; k0 += 128) {
#pragma unroll
        for (int i = 0; i < 8; ++i) {
            int idx = tid + i * 256;
            int row = idx >> 7, col = idx & 127;
            sx[row][col] = x[(size_t)(m0 + row) * D_K + k0 + col];
            sa[row][col] = loraA[(size_t)row * D_K + k0 + col];
        }
        __syncthreads();
#pragma unroll 8
        for (int k = 0; k < 128; ++k)
            acc += sx[mi][k] * sa[r][k];
        __syncthreads();
    }
    xa[(size_t)(m0 + mi) * RANK + r] = acc;
}

// -------- stage 2a: async-pipelined GEMM over pre-converted bf16 --------
__global__ __launch_bounds__(NTHREADS) void lora_gemm_async_kernel(
    const __bf16* __restrict__ xb, const __bf16* __restrict__ wb,
    const float* __restrict__ bias, const float* __restrict__ loraB,
    const float* __restrict__ xa, float* __restrict__ out) {
    __shared__ __bf16 sA[2][BM * PITCH];
    __shared__ __bf16 sB[2][BN * PITCH];

    int tid  = threadIdx.x;
    int lane = tid & 31;
    int wid  = tid >> 5;
    int m0 = blockIdx.y * BM;
    int n0 = blockIdx.x * BN;
    int wm = (wid >> 1) * 32;
    int wn = (wid & 1) * 64;

    v8f acc[2][4];
#pragma unroll
    for (int i = 0; i < 2; ++i)
#pragma unroll
        for (int j = 0; j < 4; ++j)
            acc[i][j] = v8f{};

    async_tile(xb, m0, 0, sA[0], tid);
    async_tile(wb, n0, 0, sB[0], tid);
    wait_async0();
    __syncthreads();

    constexpr int NK = D_K / BK;         // 64 iterations
    for (int kt = 0; kt < NK; ++kt) {
        int cur = kt & 1;
        if (kt + 1 < NK) {               // DMA next tile while computing current
            async_tile(xb, m0, (kt + 1) * BK, sA[cur ^ 1], tid);
            async_tile(wb, n0, (kt + 1) * BK, sB[cur ^ 1], tid);
        }
#pragma unroll
        for (int ks = 0; ks < BK; ks += 32) {
            v16bf a0 = load_fragA(sA[cur], wm,      ks, lane);
            v16bf a1 = load_fragA(sA[cur], wm + 16, ks, lane);
#pragma unroll
            for (int tj = 0; tj < 4; ++tj) {
                v16bf b = load_fragB(sB[cur], wn + tj * 16, ks, lane);
                acc[0][tj] = __builtin_amdgcn_wmma_f32_16x16x32_bf16(
                    false, a0, false, b, (short)0, acc[0][tj], false, false);
                acc[1][tj] = __builtin_amdgcn_wmma_f32_16x16x32_bf16(
                    false, a1, false, b, (short)0, acc[1][tj], false, false);
            }
        }
        wait_async0();
        __syncthreads();
    }

    epilogue(acc, m0, n0, wm, wn, lane, bias, loraB, xa, out);
}

// -------- stage 2b: fallback fused GEMM (fp32->bf16 convert in-loop) --------
__global__ __launch_bounds__(NTHREADS) void lora_gemm_fused_kernel(
    const float* __restrict__ x, const float* __restrict__ w,
    const float* __restrict__ bias, const float* __restrict__ loraB,
    const float* __restrict__ xa, float* __restrict__ out) {
    __shared__ __bf16 sA[2][BM * PITCH];
    __shared__ __bf16 sB[2][BN * PITCH];

    int tid  = threadIdx.x;
    int lane = tid & 31;
    int wid  = tid >> 5;
    int m0 = blockIdx.y * BM;
    int n0 = blockIdx.x * BN;
    int wm = (wid >> 1) * 32;
    int wn = (wid & 1) * 64;

    v8f acc[2][4];
#pragma unroll
    for (int i = 0; i < 2; ++i)
#pragma unroll
        for (int j = 0; j < 4; ++j)
            acc[i][j] = v8f{};

    f32x4 xs[8], wstg[8];
    load_stage(x, m0, 0, xs, tid);
    load_stage(w, n0, 0, wstg, tid);
    store_stage(sA[0], xs, tid);
    store_stage(sB[0], wstg, tid);
    __syncthreads();

    constexpr int NK = D_K / BK;
    for (int kt = 0; kt < NK; ++kt) {
        int cur = kt & 1;
        if (kt + 1 < NK) {
            load_stage(x, m0, (kt + 1) * BK, xs, tid);
            load_stage(w, n0, (kt + 1) * BK, wstg, tid);
        }
#pragma unroll
        for (int ks = 0; ks < BK; ks += 32) {
            v16bf a0 = load_fragA(sA[cur], wm,      ks, lane);
            v16bf a1 = load_fragA(sA[cur], wm + 16, ks, lane);
#pragma unroll
            for (int tj = 0; tj < 4; ++tj) {
                v16bf b = load_fragB(sB[cur], wn + tj * 16, ks, lane);
                acc[0][tj] = __builtin_amdgcn_wmma_f32_16x16x32_bf16(
                    false, a0, false, b, (short)0, acc[0][tj], false, false);
                acc[1][tj] = __builtin_amdgcn_wmma_f32_16x16x32_bf16(
                    false, a1, false, b, (short)0, acc[1][tj], false, false);
            }
        }
        if (kt + 1 < NK) {
            store_stage(sA[cur ^ 1], xs, tid);
            store_stage(sB[cur ^ 1], wstg, tid);
        }
        __syncthreads();
    }

    epilogue(acc, m0, n0, wm, wn, lane, bias, loraB, xa, out);
}

extern "C" void kernel_launch(void* const* d_in, const int* in_sizes, int n_in,
                              void* d_out, int out_size, void* d_ws, size_t ws_size,
                              hipStream_t stream) {
    (void)n_in; (void)out_size;
    const float* x      = (const float*)d_in[0];
    const float* weight = (const float*)d_in[1];
    const float* bias   = (const float*)d_in[2];
    const float* lora_A = (const float*)d_in[3];
    const float* lora_B = (const float*)d_in[4];
    float* out = (float*)d_out;

    int M = in_sizes[0] / D_K;                 // 8192 (B*S)

    // workspace layout: [XA fp32][X bf16][W bf16]
    char*  ws    = (char*)d_ws;
    size_t xa_b  = (size_t)M * RANK * sizeof(float);
    size_t off_x = (xa_b + 255) & ~(size_t)255;
    size_t xb_b  = (size_t)M * D_K * 2;
    size_t off_w = (off_x + xb_b + 255) & ~(size_t)255;
    size_t wb_b  = (size_t)D_N * D_K * 2;
    bool use_async = ws_size >= off_w + wb_b;

    float*  xa = (float*)ws;
    __bf16* xb = (__bf16*)(ws + off_x);
    __bf16* wb = (__bf16*)(ws + off_w);

    lora_xa_kernel<<<M / 16, 256, 0, stream>>>(x, lora_A, xa);

    dim3 grid(D_N / BN, M / BM);               // 32 x 64 blocks
    if (use_async) {
        int nx4 = M * (D_K / 4);
        int nw4 = D_N * (D_K / 4);
        cvt_bf16_kernel<<<nx4 / 256, 256, 0, stream>>>(x, (u32x2*)xb, nx4);
        cvt_bf16_kernel<<<nw4 / 256, 256, 0, stream>>>(weight, (u32x2*)wb, nw4);
        lora_gemm_async_kernel<<<grid, NTHREADS, 0, stream>>>(xb, wb, bias, lora_B, xa, out);
    } else {
        lora_gemm_fused_kernel<<<grid, NTHREADS, 0, stream>>>(x, weight, bias, lora_B, xa, out);
    }
}
